// MultiHeadAttention_67662914781416
// MI455X (gfx1250) — matrix operand order repaired and checked by reading, not compiled, not run on hardware
//
#include <hip/hip_runtime.h>
#include <hip/hip_bf16.h>

typedef __bf16 bf16_t;
typedef __attribute__((ext_vector_type(16))) __bf16 v16bf;
typedef __attribute__((ext_vector_type(8)))  __bf16 v8bf;
typedef __attribute__((ext_vector_type(8)))  float  v8f;

#define B_     4
#define T_     512
#define D_     1024
#define H_     16
#define HD_    64
#define PAST_  3584
#define CACHE_ 4096
#define NROWS_ (B_*T_)      // 2048
#define SCALE_ 0.125f       // 1/sqrt(64)

// A-fragment (16-bit A 16x32): lane holds two contiguous 8-bf16 runs:
//   k in [g*8, g*8+8) and [16+g*8, 16+g*8+8)
__device__ __forceinline__ v16bf load_a_frag(const bf16_t* base /* = row + k0 + g*8 */) {
    v8bf lo = *(const v8bf*)base;
    v8bf hi = *(const v8bf*)(base + 16);
    return __builtin_shufflevector(lo, hi, 0,1,2,3,4,5,6,7,8,9,10,11,12,13,14,15);
}
// B-fragment (16-bit B 32x16): lane holds two contiguous 8-bf16 runs:
//   k in [g*8, g*8+8) and [16+g*8, 16+g*8+8)
__device__ __forceinline__ v16bf load_b_frag(const bf16_t* base /* = col + k0 + g*8 */) {
    v8bf lo = *(const v8bf*)base;
    v8bf hi = *(const v8bf*)(base + 16);
    return __builtin_shufflevector(lo, hi, 0,1,2,3,4,5,6,7,8,9,10,11,12,13,14,15);
}

// ---------------- cast kernels ----------------
__global__ void cast_f32_bf16(const float* __restrict__ src, bf16_t* __restrict__ dst, int n) {
    int i = blockIdx.x * blockDim.x + threadIdx.x;
    int stride = gridDim.x * blockDim.x;
    for (; i < n; i += stride) dst[i] = (bf16_t)src[i];
}

// K cache: src [BH,PAST,HD] f32 -> dst [BH,CACHE,HD] bf16 (positions 0..PAST-1)
__global__ void cast_k_bf16(const float* __restrict__ src, bf16_t* __restrict__ dst, int n) {
    int i = blockIdx.x * blockDim.x + threadIdx.x;
    int stride = gridDim.x * blockDim.x;
    for (; i < n; i += stride) {
        int dd   = i & (HD_ - 1);
        int rest = i >> 6;            // bh*PAST + pos
        int pos  = rest % PAST_;
        int bh   = rest / PAST_;
        dst[((long)bh * CACHE_ + pos) * HD_ + dd] = (bf16_t)src[i];
    }
}

// V cache transpose: src [BH,PAST,HD] f32 -> dst [BH,HD,CACHE] bf16
__global__ __launch_bounds__(256) void cast_v_transpose(
    const float* __restrict__ src, bf16_t* __restrict__ dst) {
    __shared__ bf16_t tile[64][65];   // +1 pad: no bank conflicts
    int bh = blockIdx.y;              // 0..63
    int p0 = blockIdx.x * 64;         // position tile (56 tiles over PAST)
#pragma unroll
    for (int it = 0; it < 16; ++it) {
        int idx = it * 256 + threadIdx.x;
        int pr = idx >> 6, dd = idx & 63;
        tile[pr][dd] = (bf16_t)src[((long)bh * PAST_ + p0 + pr) * HD_ + dd];
    }
    __syncthreads();
#pragma unroll
    for (int it = 0; it < 16; ++it) {
        int idx = it * 256 + threadIdx.x;
        int dd = idx >> 6, pr = idx & 63;
        dst[((long)bh * HD_ + dd) * CACHE_ + p0 + pr] = tile[pr][dd];
    }
}

// ---------------- QKV projection: y = x @ W^T (bf16 WMMA, f32 acc) ----------------
// one wave per 32(M)x64(N) tile; blockIdx.y selects q/k/v
__global__ __launch_bounds__(256) void proj_qkv_wmma(
    const bf16_t* __restrict__ xb,
    const bf16_t* __restrict__ wqb,
    const bf16_t* __restrict__ wkb,
    const bf16_t* __restrict__ wvb,
    bf16_t* __restrict__ qb,
    bf16_t* __restrict__ kbuf,
    bf16_t* __restrict__ vtb) {
    int lane = threadIdx.x & 31;
    int wave = threadIdx.x >> 5;
    int tile = blockIdx.x * 8 + wave;       // 0..1023
    const int nTilesN = D_ / 64;            // 16
    int m0 = (tile / nTilesN) * 32;
    int n0 = (tile % nTilesN) * 64;
    int which = blockIdx.y;                 // 0=q 1=k 2=v
    const bf16_t* W = (which == 0) ? wqb : (which == 1) ? wkb : wvb;

    int lm = lane & 15;
    int g  = lane >> 4;

    v8f acc[2][4];
#pragma unroll
    for (int mi2 = 0; mi2 < 2; ++mi2)
#pragma unroll
        for (int f = 0; f < 4; ++f)
#pragma unroll
            for (int e = 0; e < 8; ++e) acc[mi2][f][e] = 0.f;

    const bf16_t* arow0 = xb + (long)(m0 + lm) * D_;
    const bf16_t* arow1 = xb + (long)(m0 + 16 + lm) * D_;

    for (int k0 = 0; k0 < D_; k0 += 32) {
        if (k0 + 256 < D_) __builtin_prefetch(arow0 + k0 + 256, 0, 0);
        v16bf a0 = load_a_frag(arow0 + k0 + g * 8);
        v16bf a1 = load_a_frag(arow1 + k0 + g * 8);
#pragma unroll
        for (int f = 0; f < 4; ++f) {
            int n = n0 + f * 16 + lm;
            v16bf bfrag = load_b_frag(W + (long)n * D_ + k0 + g * 8);
            acc[0][f] = __builtin_amdgcn_wmma_f32_16x16x32_bf16(
                false, a0, false, bfrag, (short)0, acc[0][f], false, false);
            acc[1][f] = __builtin_amdgcn_wmma_f32_16x16x32_bf16(
                false, a1, false, bfrag, (short)0, acc[1][f], false, false);
        }
    }

    // epilogue: scatter into head-split layouts
#pragma unroll
    for (int mi2 = 0; mi2 < 2; ++mi2)
#pragma unroll
        for (int f = 0; f < 4; ++f)
#pragma unroll
            for (int r = 0; r < 8; ++r) {
                int m   = mi2 * 16 + r + 8 * g;
                int row = m0 + m;                 // b*T + t
                int col = n0 + f * 16 + lm;       // h*HD + d
                int b   = row / T_, t = row % T_;
                int hh  = col / HD_, dd = col % HD_;
                bf16_t val = (bf16_t)acc[mi2][f][r];
                if (which == 0)
                    qb[(((long)(b * H_ + hh)) * T_ + t) * HD_ + dd] = val;
                else if (which == 1)
                    kbuf[(((long)(b * H_ + hh)) * CACHE_ + PAST_ + t) * HD_ + dd] = val;
                else
                    vtb[(((long)(b * H_ + hh)) * HD_ + dd) * CACHE_ + PAST_ + t] = val;
            }
}

// ---------------- flash attention: one wave per (b, head, 16-query tile) ----------------
__global__ __launch_bounds__(32) void attn_flash_wmma(
    const bf16_t* __restrict__ qb,
    const bf16_t* __restrict__ kbuf,
    const bf16_t* __restrict__ vtb,
    bf16_t* __restrict__ aob) {
    __shared__ bf16_t ldsP[16 * 32];        // P tile bounce: C-layout -> A-layout

    int lane = threadIdx.x;
    int qt   = blockIdx.x;                  // 0..31
    int head = blockIdx.y;
    int b    = blockIdx.z;
    int q0   = qt * 16;
    int lm   = lane & 15;
    int g    = lane >> 4;

    const bf16_t* Qp = qb   + (((long)(b * H_ + head)) * T_ + q0) * HD_;
    const bf16_t* Kp = kbuf + ((long)(b * H_ + head)) * CACHE_ * HD_;
    const bf16_t* Vt = vtb  + ((long)(b * H_ + head)) * HD_ * CACHE_;

    // Q A-fragments (K=64 -> two 16x32 frags), vectorized b128 loads
    v16bf qa[2];
#pragma unroll
    for (int kk = 0; kk < 2; ++kk)
        qa[kk] = load_a_frag(Qp + lm * HD_ + kk * 32 + g * 8);

    v8f o[4];
#pragma unroll
    for (int f = 0; f < 4; ++f)
#pragma unroll
        for (int e = 0; e < 8; ++e) o[f][e] = 0.f;

    float mi[8], li[8];
#pragma unroll
    for (int r = 0; r < 8; ++r) { mi[r] = -1e30f; li[r] = 0.f; }

    int keys = PAST_ + q0 + 16;             // keys visible to this query tile
    int nkb  = (keys + 31) >> 5;            // key blocks of 32

    for (int kbk = 0; kbk < nkb; ++kbk) {
        int kbase = kbk * 32;
        if (kbk + 1 < nkb) {
            __builtin_prefetch(Kp + (long)(kbase + 32 + lm) * HD_, 0, 0);
            __builtin_prefetch(Vt + (long)lm * CACHE_ + kbase + 32, 0, 0);
        }

        // ---- S = Q @ K^T for two key-halves (each 16 keys, K-dim 64) ----
        v8f s[2];
#pragma unroll
        for (int khf = 0; khf < 2; ++khf) {
#pragma unroll
            for (int e = 0; e < 8; ++e) s[khf][e] = 0.f;
            int key = kbase + khf * 16 + lm;
#pragma unroll
            for (int dh = 0; dh < 2; ++dh) {
                v16bf bfrag = load_b_frag(Kp + (long)key * HD_ + dh * 32 + g * 8);
                s[khf] = __builtin_amdgcn_wmma_f32_16x16x32_bf16(
                    false, qa[dh], false, bfrag, (short)0, s[khf], false, false);
            }
        }

        // ---- scale + offset-causal mask + online softmax (rows m = r + 8g) ----
        float alpha[8];
#pragma unroll
        for (int r = 0; r < 8; ++r) {
            int m   = r + 8 * g;
            int lim = PAST_ + q0 + m;       // query i sees keys j <= past_len + i
            float v0 = s[0][r] * SCALE_;
            float v1 = s[1][r] * SCALE_;
            if (kbase + lm > lim)      v0 = -1e30f;
            if (kbase + 16 + lm > lim) v1 = -1e30f;
            float t = fmaxf(v0, v1);
#pragma unroll
            for (int msk = 1; msk < 16; msk <<= 1)
                t = fmaxf(t, __shfl_xor(t, msk, 32));
            float newm = fmaxf(mi[r], t);
            float a    = __expf(mi[r] - newm);
            float p0   = __expf(v0 - newm);
            float p1   = __expf(v1 - newm);
            float rs   = p0 + p1;
#pragma unroll
            for (int msk = 1; msk < 16; msk <<= 1)
                rs += __shfl_xor(rs, msk, 32);
            li[r]    = li[r] * a + rs;
            mi[r]    = newm;
            alpha[r] = a;
            s[0][r]  = p0;
            s[1][r]  = p1;
        }
#pragma unroll
        for (int f = 0; f < 4; ++f)
#pragma unroll
            for (int r = 0; r < 8; ++r) o[f][r] *= alpha[r];

        // ---- P: C-layout -> A-layout via LDS (b16 stores, b128 reads) ----
        __syncthreads();
#pragma unroll
        for (int r = 0; r < 8; ++r) {
            int m = r + 8 * g;
            ldsP[m * 32 + lm]      = (bf16_t)s[0][r];
            ldsP[m * 32 + 16 + lm] = (bf16_t)s[1][r];
        }
        __syncthreads();
        v16bf pa = load_a_frag(&ldsP[lm * 32 + g * 8]);

        // ---- O += P @ V (4 hd tiles of 16); V transposed -> contiguous B frags ----
#pragma unroll
        for (int f = 0; f < 4; ++f) {
            v16bf vfrag = load_b_frag(Vt + (long)(f * 16 + lm) * CACHE_ + kbase + g * 8);
            o[f] = __builtin_amdgcn_wmma_f32_16x16x32_bf16(
                false, pa, false, vfrag, (short)0, o[f], false, false);
        }
    }

    // ---- normalize + store merged-head bf16 [B*T, D] ----
#pragma unroll
    for (int f = 0; f < 4; ++f)
#pragma unroll
        for (int r = 0; r < 8; ++r) {
            int m = r + 8 * g;
            float val = o[f][r] / li[r];
            aob[((long)(b * T_ + q0 + m)) * D_ + head * HD_ + f * 16 + lm] = (bf16_t)val;
        }
}

// ---------------- output projection: out = ao @ Wo^T (f32 out) ----------------
__global__ __launch_bounds__(256) void proj_out_wmma(
    const bf16_t* __restrict__ aob,
    const bf16_t* __restrict__ wob,
    float* __restrict__ out) {
    int lane = threadIdx.x & 31;
    int wave = threadIdx.x >> 5;
    int tile = blockIdx.x * 8 + wave;       // 0..1023
    const int nTilesN = D_ / 64;
    int m0 = (tile / nTilesN) * 32;
    int n0 = (tile % nTilesN) * 64;
    int lm = lane & 15;
    int g  = lane >> 4;

    v8f acc[2][4];
#pragma unroll
    for (int mi2 = 0; mi2 < 2; ++mi2)
#pragma unroll
        for (int f = 0; f < 4; ++f)
#pragma unroll
            for (int e = 0; e < 8; ++e) acc[mi2][f][e] = 0.f;

    const bf16_t* arow0 = aob + (long)(m0 + lm) * D_;
    const bf16_t* arow1 = aob + (long)(m0 + 16 + lm) * D_;

    for (int k0 = 0; k0 < D_; k0 += 32) {
        if (k0 + 256 < D_) __builtin_prefetch(arow0 + k0 + 256, 0, 0);
        v16bf a0 = load_a_frag(arow0 + k0 + g * 8);
        v16bf a1 = load_a_frag(arow1 + k0 + g * 8);
#pragma unroll
        for (int f = 0; f < 4; ++f) {
            int n = n0 + f * 16 + lm;
            v16bf bfrag = load_b_frag(wob + (long)n * D_ + k0 + g * 8);
            acc[0][f] = __builtin_amdgcn_wmma_f32_16x16x32_bf16(
                false, a0, false, bfrag, (short)0, acc[0][f], false, false);
            acc[1][f] = __builtin_amdgcn_wmma_f32_16x16x32_bf16(
                false, a1, false, bfrag, (short)0, acc[1][f], false, false);
        }
    }
#pragma unroll
    for (int mi2 = 0; mi2 < 2; ++mi2)
#pragma unroll
        for (int f = 0; f < 4; ++f)
#pragma unroll
            for (int r = 0; r < 8; ++r) {
                int m = mi2 * 16 + r + 8 * g;
                out[(long)(m0 + m) * D_ + n0 + f * 16 + lm] = acc[mi2][f][r];
            }
}

// ---------------- host launcher ----------------
extern "C" void kernel_launch(void* const* d_in, const int* in_sizes, int n_in,
                              void* d_out, int out_size, void* d_ws, size_t ws_size,
                              hipStream_t stream) {
    (void)in_sizes; (void)n_in; (void)out_size; (void)ws_size;
    const float* x      = (const float*)d_in[0];
    const float* k_prev = (const float*)d_in[1];
    const float* v_prev = (const float*)d_in[2];
    const float* Wq     = (const float*)d_in[3];
    const float* Wk     = (const float*)d_in[4];
    const float* Wv     = (const float*)d_in[5];
    const float* Wo     = (const float*)d_in[6];

    char* ws = (char*)d_ws;
    size_t off = 0;
    auto alloc = [&](size_t bytes) -> void* {
        void* p = ws + off;
        off += (bytes + 255) & ~(size_t)255;
        return p;
    };
    bf16_t* xb   = (bf16_t*)alloc((size_t)NROWS_ * D_ * 2);            // 4 MiB
    bf16_t* wqb  = (bf16_t*)alloc((size_t)D_ * D_ * 2);                // 2 MiB
    bf16_t* wkb  = (bf16_t*)alloc((size_t)D_ * D_ * 2);
    bf16_t* wvb  = (bf16_t*)alloc((size_t)D_ * D_ * 2);
    bf16_t* wob  = (bf16_t*)alloc((size_t)D_ * D_ * 2);
    bf16_t* qb   = (bf16_t*)alloc((size_t)B_ * H_ * T_ * HD_ * 2);     // 4 MiB
    bf16_t* kbuf = (bf16_t*)alloc((size_t)B_ * H_ * CACHE_ * HD_ * 2); // 32 MiB
    bf16_t* vtb  = (bf16_t*)alloc((size_t)B_ * H_ * HD_ * CACHE_ * 2); // 32 MiB, transposed
    bf16_t* aob  = (bf16_t*)alloc((size_t)NROWS_ * D_ * 2);            // 4 MiB

    // 1) casts (+ V transpose)
    cast_f32_bf16<<<2048, 256, 0, stream>>>(x, xb, NROWS_ * D_);
    cast_f32_bf16<<<1024, 256, 0, stream>>>(Wq, wqb, D_ * D_);
    cast_f32_bf16<<<1024, 256, 0, stream>>>(Wk, wkb, D_ * D_);
    cast_f32_bf16<<<1024, 256, 0, stream>>>(Wv, wvb, D_ * D_);
    cast_f32_bf16<<<1024, 256, 0, stream>>>(Wo, wob, D_ * D_);
    cast_k_bf16<<<4096, 256, 0, stream>>>(k_prev, kbuf, B_ * H_ * PAST_ * HD_);
    cast_v_transpose<<<dim3(PAST_ / 64, B_ * H_), 256, 0, stream>>>(v_prev, vtb);

    // 2) QKV projections (q -> qb; k_new at cache offset PAST; v_new transposed)
    proj_qkv_wmma<<<dim3(128, 3), 256, 0, stream>>>(xb, wqb, wkb, wvb, qb, kbuf, vtb);

    // 3) flash attention over 4096-entry cache with past-offset causal mask
    attn_flash_wmma<<<dim3(32, H_, B_), 32, 0, stream>>>(qb, kbuf, vtb, aob);

    // 4) output projection -> f32 result
    proj_out_wmma<<<128, 256, 0, stream>>>(aob, wob, (float*)d_out);
}
